// DLRM_Net_8022998909721
// MI455X (gfx1250) — matrix operand order repaired and checked by reading, not compiled, not run on hardware
//
#include <hip/hip_runtime.h>
#include <hip/hip_bf16.h>
#include <math.h>

typedef __attribute__((ext_vector_type(16))) _Float16 v16h;
typedef __attribute__((ext_vector_type(8)))  _Float16 v8h;
typedef __attribute__((ext_vector_type(8)))  float    v8f;

#define BATCH   16384
#define NTAB    26
#define NFEAT   27            // 1 dense + 26 embeddings
#define NROWS   200000
#define DIM     64
#define TROW    (NFEAT * DIM) // 1728 halves per sample in T buffer
#define RCOLS   416           // 64 + 351 tril + 1 pad

// ---------------------------------------------------------------------------
// f32 -> f16 convert with zero padding (rows/cols beyond src are zeroed)
// ---------------------------------------------------------------------------
__global__ void cvt_pad_f16(const float* __restrict__ src, _Float16* __restrict__ dst,
                            int srcR, int srcC, int dstR, int dstC) {
  int gid = blockIdx.x * blockDim.x + threadIdx.x;
  if (gid >= dstR * dstC) return;
  int r = gid / dstC, c = gid % dstC;
  float v = (r < srcR && c < srcC) ? src[(size_t)r * srcC + c] : 0.0f;
  dst[gid] = (_Float16)v;
}

// ---------------------------------------------------------------------------
// Embedding gather (bags are singletons): T[b][t+1][k] = f16(emb[t][idx[t][b]][k])
// ---------------------------------------------------------------------------
__global__ void gather_embed(const float* __restrict__ emb, const int* __restrict__ idx,
                             _Float16* __restrict__ Tbuf) {
  size_t gid = (size_t)blockIdx.x * blockDim.x + threadIdx.x;
  int    k   = (int)(gid & (DIM - 1));
  size_t tb  = gid >> 6;
  int    b   = (int)(tb % BATCH);
  int    t   = (int)(tb / BATCH);
  if (t >= NTAB) return;
  int row = idx[t * BATCH + b];
  float v = emb[((size_t)t * NROWS + row) * DIM + k];
  Tbuf[(size_t)b * TROW + (size_t)(t + 1) * DIM + k] = (_Float16)v;
}

// ---------------------------------------------------------------------------
// WMMA fragment loaders (CDNA5 16x16x32 f16 layouts, wave32)
//   A 16x32: lane holds row (lane&15); halves 0-7 at K=(lane>>4)*8, 8-15 at +16
//   B 32x16: lane holds col (lane&15); halves 0-7 at K=(lane>>4)*8, 8-15 at +16
// ---------------------------------------------------------------------------
__device__ __forceinline__ v16h load_a_frag(const _Float16* __restrict__ A, int lda,
                                            int rowBase, int k0, int lane) {
  const _Float16* p = A + (size_t)(rowBase + (lane & 15)) * lda + k0 + ((lane >> 4) << 3);
  v8h lo = *(const v8h*)(p);
  v8h hi = *(const v8h*)(p + 16);
  v16h a;
#pragma unroll
  for (int i = 0; i < 8; ++i) { a[i] = lo[i]; a[i + 8] = hi[i]; }
  return a;
}

__device__ __forceinline__ v16h load_b_frag(const _Float16* __restrict__ W, int ldw,
                                            int colBase, int k0, int lane) {
  const _Float16* p = W + (size_t)(colBase + (lane & 15)) * ldw + k0 + ((lane >> 4) << 3);
  v8h lo = *(const v8h*)(p);
  v8h hi = *(const v8h*)(p + 16);
  v16h b;
#pragma unroll
  for (int i = 0; i < 8; ++i) { b[i] = lo[i]; b[i + 8] = hi[i]; }
  return b;
}

// ---------------------------------------------------------------------------
// Generic GEMM:  out[M,N] = act( A[M,K](f16) @ W[N,K]^T(f16) + bias )
// Register-blocked: one wave computes a 32x32 output block (2x2 WMMA tiles,
// 4 independent accumulator chains; 4 WMMAs per 8 b128 operand loads).
// mode 0: ReLU, store f16 at out[m*ldout + n]
// mode 1: sigmoid, store f32 at outF[m] (column 0 only; N padded to 32)
// ---------------------------------------------------------------------------
__global__ __launch_bounds__(256)
void wmma_gemm_bias_act(const _Float16* __restrict__ A, int lda,
                        const _Float16* __restrict__ W, int ldw,
                        const float* __restrict__ bias,
                        _Float16* __restrict__ outH, float* __restrict__ outF,
                        int ldout, int M, int N, int K, int mode) {
  const int lane   = threadIdx.x & 31;
  const int tilesN = N >> 5;                    // 32-wide blocks
  const int blk    = blockIdx.x * 8 + (threadIdx.x >> 5);
  if (blk >= (M >> 5) * tilesN) return;         // wave-uniform guard
  const int bm = (blk / tilesN) << 5;           // row base
  const int bn = (blk % tilesN) << 5;           // col base

  v8f acc00 = {}, acc01 = {}, acc10 = {}, acc11 = {};
  for (int k0 = 0; k0 < K; k0 += 32) {
    v16h a0 = load_a_frag(A, lda, bm,      k0, lane);
    v16h a1 = load_a_frag(A, lda, bm + 16, k0, lane);
    v16h b0 = load_b_frag(W, ldw, bn,      k0, lane);
    v16h b1 = load_b_frag(W, ldw, bn + 16, k0, lane);
    acc00 = __builtin_amdgcn_wmma_f32_16x16x32_f16(false, a0, false, b0, (short)0, acc00, false, false);
    acc01 = __builtin_amdgcn_wmma_f32_16x16x32_f16(false, a0, false, b1, (short)0, acc01, false, false);
    acc10 = __builtin_amdgcn_wmma_f32_16x16x32_f16(false, a1, false, b0, (short)0, acc10, false, false);
    acc11 = __builtin_amdgcn_wmma_f32_16x16x32_f16(false, a1, false, b1, (short)0, acc11, false, false);
  }

  const int nloc  = lane & 15;
  const int mloc  = (lane >> 4) << 3;
  if (mode == 0) {
    float bv0 = bias[bn + nloc];
    float bv1 = bias[bn + 16 + nloc];
#pragma unroll
    for (int r = 0; r < 8; ++r) {
      int m0 = bm + mloc + r, m1 = bm + 16 + mloc + r;
      float v;
      v = acc00[r] + bv0; v = v > 0.0f ? v : 0.0f;
      outH[(size_t)m0 * ldout + bn + nloc]      = (_Float16)v;
      v = acc01[r] + bv1; v = v > 0.0f ? v : 0.0f;
      outH[(size_t)m0 * ldout + bn + 16 + nloc] = (_Float16)v;
      v = acc10[r] + bv0; v = v > 0.0f ? v : 0.0f;
      outH[(size_t)m1 * ldout + bn + nloc]      = (_Float16)v;
      v = acc11[r] + bv1; v = v > 0.0f ? v : 0.0f;
      outH[(size_t)m1 * ldout + bn + 16 + nloc] = (_Float16)v;
    }
  } else {
    if (nloc == 0) {                            // only real column 0 (bn==0)
      float bv = bias[0];
#pragma unroll
      for (int r = 0; r < 8; ++r) {
        float v0 = acc00[r] + bv;
        float v1 = acc10[r] + bv;
        outF[bm + mloc + r]      = 1.0f / (1.0f + __expf(-v0));
        outF[bm + 16 + mloc + r] = 1.0f / (1.0f + __expf(-v1));
      }
    }
  }
}

// ---------------------------------------------------------------------------
// Interaction: one wave per sample. T=[27,64] padded to [32,64].
// Z = T @ T^T via 16x16x32 WMMA; tile (tm=0,tn=1) has no i>j elems -> skipped.
// Writes R[b] = [ x(64 f16) | tril(Z) (351 f16) | 0 pad ]
// ---------------------------------------------------------------------------
__global__ __launch_bounds__(256)
void interact_wmma(const _Float16* __restrict__ Tbuf, _Float16* __restrict__ R) {
  const int lane = threadIdx.x & 31;
  const int b    = blockIdx.x * 8 + (threadIdx.x >> 5);
  const _Float16* T = Tbuf + (size_t)b * TROW;
  _Float16* r = R + (size_t)b * RCOLS;

  // copy dense features (T row 0) into R[0:64]; zero the pad column
  r[lane]      = T[lane];
  r[lane + 32] = T[lane + 32];
  if (lane == 0) r[RCOLS - 1] = (_Float16)0.0f;

  // A fragments: rows >= 27 zero
  v16h a1[2], a0[2], bm0[2], bm1[2];
#pragma unroll
  for (int kc = 0; kc < 2; ++kc) {
#pragma unroll
    for (int tm = 0; tm < 2; ++tm) {
      int m  = tm * 16 + (lane & 15);
      int kb = kc * 32 + ((lane >> 4) << 3);
      v16h f = {};
      if (m < NFEAT) {
        const _Float16* p = T + (size_t)m * DIM + kb;
        v8h lo = *(const v8h*)(p);
        v8h hi = *(const v8h*)(p + 16);
#pragma unroll
        for (int i = 0; i < 8; ++i) { f[i] = lo[i]; f[i + 8] = hi[i]; }
      }
      if (tm == 0) a0[kc] = f; else a1[kc] = f;
    }
#pragma unroll
    for (int tn = 0; tn < 2; ++tn) {
      int n  = tn * 16 + (lane & 15);
      int kb = kc * 32 + ((lane >> 4) << 3);
      v16h f = {};
      if (n < NFEAT) {
        const _Float16* p = T + (size_t)n * DIM + kb;
        v8h lo = *(const v8h*)(p);
        v8h hi = *(const v8h*)(p + 16);
#pragma unroll
        for (int i = 0; i < 8; ++i) { f[i] = lo[i]; f[i + 8] = hi[i]; }
      }
      if (tn == 0) bm0[kc] = f; else bm1[kc] = f;
    }
  }

  v8f acc00 = {}, acc10 = {}, acc11 = {};
  acc00 = __builtin_amdgcn_wmma_f32_16x16x32_f16(false, a0[0], false, bm0[0], (short)0, acc00, false, false);
  acc10 = __builtin_amdgcn_wmma_f32_16x16x32_f16(false, a1[0], false, bm0[0], (short)0, acc10, false, false);
  acc11 = __builtin_amdgcn_wmma_f32_16x16x32_f16(false, a1[0], false, bm1[0], (short)0, acc11, false, false);
  acc00 = __builtin_amdgcn_wmma_f32_16x16x32_f16(false, a0[1], false, bm0[1], (short)0, acc00, false, false);
  acc10 = __builtin_amdgcn_wmma_f32_16x16x32_f16(false, a1[1], false, bm0[1], (short)0, acc10, false, false);
  acc11 = __builtin_amdgcn_wmma_f32_16x16x32_f16(false, a1[1], false, bm1[1], (short)0, acc11, false, false);

  // scatter lower triangle: flat(i,j) = i*(i-1)/2 + j  (i>j, i<27)
  const int nloc = lane & 15;
  const int mloc = (lane >> 4) << 3;
#pragma unroll
  for (int rr = 0; rr < 8; ++rr) {
    { int i = mloc + rr,      j = nloc;
      if (i > j)                 r[64 + i * (i - 1) / 2 + j] = (_Float16)acc00[rr]; }
    { int i = 16 + mloc + rr, j = nloc;               // i>=16 > j<=15 always
      if (i < NFEAT)             r[64 + i * (i - 1) / 2 + j] = (_Float16)acc10[rr]; }
    { int i = 16 + mloc + rr, j = 16 + nloc;
      if (i < NFEAT && i > j)    r[64 + i * (i - 1) / 2 + j] = (_Float16)acc11[rr]; }
  }
}

// ---------------------------------------------------------------------------
extern "C" void kernel_launch(void* const* d_in, const int* in_sizes, int n_in,
                              void* d_out, int out_size, void* d_ws, size_t ws_size,
                              hipStream_t stream) {
  (void)in_sizes; (void)n_in; (void)out_size; (void)ws_size;
  const float* dense_x = (const float*)d_in[0];
  const int*   lS_i    = (const int*)d_in[2];     // lS_o (d_in[1]) unused: bags are singletons
  const float* emb     = (const float*)d_in[3];
  const float* bw0 = (const float*)d_in[4];  const float* bb0 = (const float*)d_in[5];
  const float* bw1 = (const float*)d_in[6];  const float* bb1 = (const float*)d_in[7];
  const float* bw2 = (const float*)d_in[8];  const float* bb2 = (const float*)d_in[9];
  const float* tw0 = (const float*)d_in[10]; const float* tb0 = (const float*)d_in[11];
  const float* tw1 = (const float*)d_in[12]; const float* tb1 = (const float*)d_in[13];
  const float* tw2 = (const float*)d_in[14]; const float* tb2 = (const float*)d_in[15];
  float* out = (float*)d_out;

  char* ws = (char*)d_ws;
  size_t off = 0;
  auto alloc = [&](size_t bytes) -> char* {
    char* p = ws + off; off += (bytes + 255) & ~(size_t)255; return p;
  };
  _Float16* A0   = (_Float16*)alloc((size_t)BATCH * 32 * 2);   // dense_x padded 13->32
  _Float16* W0h  = (_Float16*)alloc((size_t)512 * 32 * 2);
  _Float16* W1h  = (_Float16*)alloc((size_t)256 * 512 * 2);
  _Float16* W2h  = (_Float16*)alloc((size_t)64 * 256 * 2);
  _Float16* TW0h = (_Float16*)alloc((size_t)512 * RCOLS * 2);  // 415 -> 416
  _Float16* TW1h = (_Float16*)alloc((size_t)256 * 512 * 2);
  _Float16* TW2h = (_Float16*)alloc((size_t)32 * 256 * 2);     // 1 -> 32 rows
  _Float16* X1   = (_Float16*)alloc((size_t)BATCH * 512 * 2);  // reused as top-Z1
  _Float16* X2   = (_Float16*)alloc((size_t)BATCH * 256 * 2);  // reused as top-Z2
  _Float16* Tbuf = (_Float16*)alloc((size_t)BATCH * TROW * 2); // [B][27][64]
  _Float16* R    = (_Float16*)alloc((size_t)BATCH * RCOLS * 2);

  auto cvt = [&](const float* s, _Float16* d, int sr, int sc, int dr, int dc) {
    int total = dr * dc;
    cvt_pad_f16<<<dim3((total + 255) / 256), dim3(256), 0, stream>>>(s, d, sr, sc, dr, dc);
  };
  auto gemm = [&](const _Float16* A, int lda, const _Float16* W, int ldw,
                  const float* bias, _Float16* oH, float* oF, int ldo,
                  int M, int N, int K, int mode) {
    int blks = (M >> 5) * (N >> 5);             // 32x32 waves
    wmma_gemm_bias_act<<<dim3((blks + 7) / 8), dim3(256), 0, stream>>>(
        A, lda, W, ldw, bias, oH, oF, ldo, M, N, K, mode);
  };

  // 1) precision conversion / padding
  cvt(dense_x, A0,   BATCH, 13,  BATCH, 32);
  cvt(bw0,     W0h,  512,   13,  512,   32);
  cvt(bw1,     W1h,  256,  512,  256,  512);
  cvt(bw2,     W2h,  64,   256,  64,   256);
  cvt(tw0,     TW0h, 512,  415,  512,  RCOLS);
  cvt(tw1,     TW1h, 256,  512,  256,  512);
  cvt(tw2,     TW2h, 1,    256,  32,   256);

  // 2) bottom MLP (layer 3 writes f16 straight into T row 0, ldout = 1728)
  gemm(A0, 32,  W0h, 32,  bb0, X1,   nullptr, 512,  BATCH, 512, 32,  0);
  gemm(X1, 512, W1h, 512, bb1, X2,   nullptr, 256,  BATCH, 256, 512, 0);
  gemm(X2, 256, W2h, 256, bb2, Tbuf, nullptr, TROW, BATCH, 64,  256, 0);

  // 3) embedding gather into T rows 1..26
  {
    size_t total = (size_t)NTAB * BATCH * DIM;
    gather_embed<<<dim3((unsigned)((total + 255) / 256)), dim3(256), 0, stream>>>(emb, lS_i, Tbuf);
  }

  // 4) pairwise interactions -> R
  interact_wmma<<<dim3(BATCH / 8), dim3(256), 0, stream>>>(Tbuf, R);

  // 5) top MLP (final layer: sigmoid, f32 output; N padded to 32)
  gemm(R,  RCOLS, TW0h, RCOLS, tb0, X1, nullptr, 512, BATCH, 512, RCOLS, 0);
  gemm(X1, 512,   TW1h, 512,   tb1, X2, nullptr, 256, BATCH, 256, 512,   0);
  gemm(X2, 256,   TW2h, 256,   tb2, nullptr, out, 1,  BATCH, 32,  256,   1);
}
